// WideRNN_42923903156921
// MI455X (gfx1250) — matrix operand order repaired and checked by reading, not compiled, not run on hardware
//
#include <hip/hip_runtime.h>
#include <hip/hip_bf16.h>
#include <math.h>

#define NN 16
#define BB 16
#define TT 512
#define II 256
#define HH 512

typedef __attribute__((ext_vector_type(16))) __bf16 v16bf;
typedef __attribute__((ext_vector_type(8)))  __bf16 v8bf;
typedef __attribute__((ext_vector_type(8)))  float  v8f;

static __device__ inline v16bf load_cvt16(const float* __restrict__ p) {
    v16bf r;
#pragma unroll
    for (int e = 0; e < 16; ++e) r[e] = (__bf16)p[e];
    return r;
}

// ---------------------------------------------------------------------------
// Kernel A: x_proj[n,b,t,h] = x[n,b,t,:] . w_ih[n,h,:] + b_ih[n,h]
// Written into d_out's (N,B,T,H) output region (consumed+overwritten by scan).
// One wave computes a 16(M) x 128(N) strip, K = 256 in 8 steps of 32.
// ---------------------------------------------------------------------------
__global__ __launch_bounds__(256) void xproj_kernel(
    const float* __restrict__ x, const float* __restrict__ w_ih,
    const float* __restrict__ b_ih, float* __restrict__ out) {
    const int lane = threadIdx.x & 31;
    const int wave = threadIdx.x >> 5;
    const int gw   = blockIdx.x * 8 + wave;   // 32768 wave-tasks
    const int cg    = gw & 3;                 // col group of 128
    const int mtile = (gw >> 2) & 511;        // M tile (16 rows of b*T+t)
    const int n     = gw >> 11;               // net

    const int cl = lane & 15;
    const int hi = lane >> 4;                 // lane half
    const int row = mtile * 16 + cl;          // = b*T + t
    const float* xrow = x + ((size_t)n * BB * TT + row) * II;

    v8f acc[8];
#pragma unroll
    for (int i = 0; i < 8; ++i)
#pragma unroll
        for (int j = 0; j < 8; ++j) acc[i][j] = 0.0f;

#pragma unroll
    for (int kc = 0; kc < II; kc += 32) {
        // A fragment (contiguous K: lanes0-15 K=base..base+15, lanes16-31 K=base+16..base+31)
        const float* ap = xrow + kc + hi * 16;
        v16bf a;
#pragma unroll
        for (int e = 0; e < 8; ++e) {
            a[e]     = (__bf16)ap[e];
            a[e + 8] = (__bf16)ap[e + 8];
        }
#pragma unroll
        for (int ct = 0; ct < 8; ++ct) {
            const int col = cg * 128 + ct * 16 + cl;
            const float* wp = w_ih + ((size_t)n * HH + col) * II + kc + hi * 16;
            v16bf b = load_cvt16(wp);   // B[k][col] = w_ih[col][k], 16 contiguous f32
            acc[ct] = __builtin_amdgcn_wmma_f32_16x16x32_bf16(
                false, a, false, b, (short)0, acc[ct], false, false);
        }
    }
    // store + bias; C/D layout: lane half hi -> M = v + 8*hi, N = cl
#pragma unroll
    for (int ct = 0; ct < 8; ++ct) {
        const int col = cg * 128 + ct * 16 + cl;
        const float bias = b_ih[n * HH + col];
#pragma unroll
        for (int v = 0; v < 8; ++v) {
            const int m = v + hi * 8;
            out[((size_t)n * BB * TT + (size_t)(mtile * 16 + m)) * HH + col] =
                acc[ct][v] + bias;
        }
    }
}

// ---------------------------------------------------------------------------
// Kernel B: recurrent scan. One 1024-thread workgroup per net; wave w owns
// output columns [16w, 16w+16). w_hh^T slice register-resident as 16 bf16
// B-fragments. h double-buffered in LDS (bf16). One barrier per step.
// ---------------------------------------------------------------------------
__global__ __launch_bounds__(1024) void rnn_scan_kernel(
    const float* __restrict__ w_hh, const float* __restrict__ b_hh,
    float* __restrict__ out, float* __restrict__ hn) {
    const int n    = blockIdx.x;
    const int lane = threadIdx.x & 31;
    const int wave = threadIdx.x >> 5;     // 0..31 -> column tile
    const int cl   = lane & 15;
    const int hi   = lane >> 4;
    const int o    = wave * 16 + cl;       // output column owned by this lane

    // padded row stride: rows hit distinct bank groups for 16B reads
    __shared__ __align__(16) __bf16 hbuf[2][16][520];

    // Preload w_hh^T B-fragments: B[k][o] = w_hh[o][k] (contiguous rows)
    v16bf wv[16];
#pragma unroll
    for (int f = 0; f < 16; ++f) {
        const float* wp = w_hh + ((size_t)n * HH + o) * HH + f * 32 + hi * 16;
        wv[f] = load_cvt16(wp);
    }
    const float bias = b_hh[n * HH + o];

    // h0 = 0
    for (int i = threadIdx.x; i < 16 * 520; i += 1024)
        ((__bf16*)hbuf[0])[i] = (__bf16)0.0f;
    __syncthreads();

    float* xbase = out + (size_t)n * BB * TT * HH;   // x_proj in / h out

    for (int t = 0; t < TT; ++t) {
        const int cur = t & 1, nxt = cur ^ 1;

        // issue x_proj tile loads early (consumed after the WMMA chain)
        float xt[8];
#pragma unroll
        for (int v = 0; v < 8; ++v) {
            const int b = v + hi * 8;
            xt[v] = xbase[((size_t)b * TT + t) * HH + o];
        }

        v8f acc;
#pragma unroll
        for (int j = 0; j < 8; ++j) acc[j] = 0.0f;

#pragma unroll
        for (int f = 0; f < 16; ++f) {
            const int k = f * 32 + hi * 16;
            const v8bf lo = *(const v8bf*)&hbuf[cur][cl][k];
            const v8bf hh = *(const v8bf*)&hbuf[cur][cl][k + 8];
            v16bf a = __builtin_shufflevector(lo, hh,
                0, 1, 2, 3, 4, 5, 6, 7, 8, 9, 10, 11, 12, 13, 14, 15);
            acc = __builtin_amdgcn_wmma_f32_16x16x32_bf16(
                false, a, false, wv[f], (short)0, acc, false, false);
        }

#pragma unroll
        for (int v = 0; v < 8; ++v) {
            const int b = v + hi * 8;
            const float h = tanhf(xt[v] + acc[v] + bias);
            hbuf[nxt][b][o] = (__bf16)h;                     // next-step A operand
            xbase[((size_t)b * TT + t) * HH + o] = h;        // stream output
            if (t == TT - 1) hn[((size_t)n * BB + b) * HH + o] = h;
        }
        __syncthreads();   // writes to hbuf[nxt] visible before next step reads
    }
}

extern "C" void kernel_launch(void* const* d_in, const int* in_sizes, int n_in,
                              void* d_out, int out_size, void* d_ws, size_t ws_size,
                              hipStream_t stream) {
    (void)in_sizes; (void)n_in; (void)out_size; (void)d_ws; (void)ws_size;
    const float* x    = (const float*)d_in[0];
    const float* w_ih = (const float*)d_in[1];
    const float* w_hh = (const float*)d_in[2];
    const float* b_ih = (const float*)d_in[3];
    const float* b_hh = (const float*)d_in[4];
    float* out = (float*)d_out;
    float* hn  = out + (size_t)NN * BB * TT * HH;

    xproj_kernel<<<4096, 256, 0, stream>>>(x, w_ih, b_ih, out);
    rnn_scan_kernel<<<16, 1024, 0, stream>>>(w_hh, b_hh, out, hn);
}
